// DecoderBlock_80590766342872
// MI455X (gfx1250) — matrix operand order repaired and checked by reading, not compiled, not run on hardware
//
#include <hip/hip_runtime.h>
#include <hip/hip_bf16.h>

typedef __attribute__((ext_vector_type(16))) __bf16 v16bf;
typedef __attribute__((ext_vector_type(8)))  float  v8f;

#define NHEADS 16
#define DHEAD  64
#define HID    1024

// GEMM tiling
#define BM 256          // block rows (8 waves x 32 rows)
#define BN 64           // block cols
#define BK 32           // k-chunk
#define LDP 40          // padded LDS row stride in bf16 (80B, 16B aligned)

union FragBF {
    v16bf v;
    unsigned short s[16];
    uint4 u4[2];
};

__device__ __forceinline__ unsigned short f2bf(float f) {
    unsigned int u = __float_as_uint(f);
    u += 0x7FFFu + ((u >> 16) & 1u);   // round-to-nearest-even
    return (unsigned short)(u >> 16);
}

// ---------------- weight convert + transpose: w[K][N] f32 -> wt[N][K] bf16 ---
__global__ __launch_bounds__(256) void wt_kernel(const float* __restrict__ w,
                                                 unsigned short* __restrict__ wt,
                                                 int K, int N) {
    size_t tid = (size_t)blockIdx.x * 256 + threadIdx.x;
    if (tid >= (size_t)K * N) return;
    int k = (int)(tid % K);
    size_t n = tid / K;
    wt[n * K + k] = f2bf(w[(size_t)k * N + n]);
}

// ---------------- layernorm: f32 row -> bf16 row ----------------------------
__global__ __launch_bounds__(256) void ln_kernel(const float* __restrict__ x,
                                                 const float* __restrict__ w,
                                                 unsigned short* __restrict__ out) {
    int row = blockIdx.x;
    const float* xr = x + (size_t)row * HID;
    float s = 0.f, s2 = 0.f;
    for (int i = threadIdx.x; i < HID; i += 256) {
        float v = xr[i];
        s += v; s2 += v * v;
    }
    __shared__ float red0[256], red1[256];
    red0[threadIdx.x] = s; red1[threadIdx.x] = s2;
    __syncthreads();
    for (int st = 128; st > 0; st >>= 1) {
        if (threadIdx.x < (unsigned)st) {
            red0[threadIdx.x] += red0[threadIdx.x + st];
            red1[threadIdx.x] += red1[threadIdx.x + st];
        }
        __syncthreads();
    }
    float mu = red0[0] * (1.0f / HID);
    float var = red1[0] * (1.0f / HID) - mu * mu;
    float rs = rsqrtf(var + 1e-5f);
    for (int i = threadIdx.x; i < HID; i += 256)
        out[(size_t)row * HID + i] = f2bf((xr[i] - mu) * rs * w[i]);
}

// ---------------- GEMM: C = A(bf16 MxK) * Bt(bf16 NxK)^T, f32 acc -----------
// LDS-staged: A-tile BMxBK, B-tile BNxBK per block. Each wave computes 32x64.
// EPI 0: f32 out.  EPI 1: f32 out = acc + res.  EPI 2: bf16 out = gelu(acc).
template <int EPI>
__global__ __launch_bounds__(256) void gemm_bf16(const unsigned short* __restrict__ A,
                                                 const unsigned short* __restrict__ Bt,
                                                 float* __restrict__ outf,
                                                 unsigned short* __restrict__ outb,
                                                 const float* __restrict__ res,
                                                 int M, int N, int K) {
    __shared__ unsigned short As[BM * LDP];
    __shared__ unsigned short Bs[BN * LDP];

    int tid  = threadIdx.x;
    int wave = tid >> 5;
    int lane = tid & 31;
    int n16 = lane & 15, g = lane >> 4;
    int rowblk = blockIdx.y * BM;
    int col0   = blockIdx.x * BN;

    v8f acc[2][4];
#pragma unroll
    for (int sp = 0; sp < 2; sp++)
#pragma unroll
        for (int t = 0; t < 4; t++)
#pragma unroll
            for (int r = 0; r < 8; r++) acc[sp][t][r] = 0.f;

    const unsigned short* Arow = A + (size_t)(rowblk + tid) * K;
    const unsigned short* Brow = Bt + (size_t)(col0 + (tid & (BN - 1))) * K;

    for (int k0 = 0; k0 < K; k0 += BK) {
        // ---- stage A (each thread: one 32-bf16 row) and B (threads < BN) ----
        {
            const uint4* srcA = reinterpret_cast<const uint4*>(Arow + k0);
            uint4* dstA = reinterpret_cast<uint4*>(As + tid * LDP);
            dstA[0] = srcA[0]; dstA[1] = srcA[1];
            dstA[2] = srcA[2]; dstA[3] = srcA[3];
            if (tid < BN) {
                const uint4* srcB = reinterpret_cast<const uint4*>(Brow + k0);
                uint4* dstB = reinterpret_cast<uint4*>(Bs + tid * LDP);
                dstB[0] = srcB[0]; dstB[1] = srcB[1];
                dstB[2] = srcB[2]; dstB[3] = srcB[3];
            }
            if (k0 + BK < K) {                 // gfx1250 global_prefetch_b8
                __builtin_prefetch(Arow + k0 + BK, 0, 3);
                if (tid < BN) __builtin_prefetch(Brow + k0 + BK, 0, 3);
            }
        }
        __syncthreads();

        // ---- compute: wave covers rows wave*32..wave*32+31 ----
        FragBF a[2];
#pragma unroll
        for (int sp = 0; sp < 2; sp++) {
            const unsigned short* p = As + (wave * 32 + sp * 16 + n16) * LDP;
            a[sp].u4[0] = *reinterpret_cast<const uint4*>(p + g * 16);
            a[sp].u4[1] = *reinterpret_cast<const uint4*>(p + g * 16 + 8);
        }
#pragma unroll
        for (int t = 0; t < 4; t++) {
            FragBF b;
            const unsigned short* p = Bs + (t * 16 + n16) * LDP + g * 16;
            b.u4[0] = reinterpret_cast<const uint4*>(p)[0];
            b.u4[1] = reinterpret_cast<const uint4*>(p)[1];
#pragma unroll
            for (int sp = 0; sp < 2; sp++)
                acc[sp][t] = __builtin_amdgcn_wmma_f32_16x16x32_bf16(
                    false, a[sp].v, false, b.v, (short)0, acc[sp][t], false, false);
        }
        __syncthreads();
    }

    // ---- epilogue ----
#pragma unroll
    for (int sp = 0; sp < 2; sp++) {
        int row0 = rowblk + wave * 32 + sp * 16;
#pragma unroll
        for (int t = 0; t < 4; t++) {
#pragma unroll
            for (int r = 0; r < 8; r++) {
                int m = r + 8 * g;
                size_t idx = (size_t)(row0 + m) * N + col0 + t * 16 + n16;
                float v = acc[sp][t][r];
                if (EPI == 0) {
                    outf[idx] = v;
                } else if (EPI == 1) {
                    outf[idx] = v + res[idx];
                } else {
                    float gel = 0.5f * v * (1.0f + erff(v * 0.70710678118f));
                    outb[idx] = f2bf(gel);
                }
            }
        }
    }
}

// ---------------- qkv f32 (M x 3H) -> per-head bf16 --------------------------
// Q,K: (B,nH,S,64) row-major.  V: transposed (B,nH,64,S) so P@V B-frags are
// contiguous 32B per lane.
__global__ __launch_bounds__(256) void repack_qkv(const float* __restrict__ qkv,
                                                  unsigned short* __restrict__ Qb,
                                                  unsigned short* __restrict__ Kb,
                                                  unsigned short* __restrict__ Vt,
                                                  int S) {
    size_t tid = (size_t)blockIdx.x * 256 + threadIdx.x;
    size_t total = (size_t)2 * S * HID;
    if (tid >= total) return;
    int d = (int)(tid % DHEAD); size_t t = tid / DHEAD;
    int h = (int)(t % NHEADS);  t /= NHEADS;
    int s = (int)(t % S);
    int b = (int)(t / S);
    int bh = b * NHEADS + h;
    size_t src = (size_t)(b * S + s) * (3 * HID);
    size_t dqk = ((size_t)bh * S + s) * DHEAD + d;
    size_t dv  = ((size_t)bh * DHEAD + d) * S + s;
    Qb[dqk] = f2bf(qkv[src + h * DHEAD + d]);
    Kb[dqk] = f2bf(qkv[src + HID + h * DHEAD + d]);
    Vt[dv]  = f2bf(qkv[src + 2 * HID + h * DHEAD + d]);
}

// ---------------- causal flash attention, one wave per 16 query rows --------
__global__ __launch_bounds__(32) void attn_kernel(const unsigned short* __restrict__ Qb,
                                                  const unsigned short* __restrict__ Kb,
                                                  const unsigned short* __restrict__ Vt,
                                                  unsigned short* __restrict__ outb,
                                                  int S) {
    int qb = blockIdx.x;       // query block of 16 rows
    int h  = blockIdx.y;
    int b  = blockIdx.z;
    int bh = b * NHEADS + h;
    const unsigned short* Q  = Qb + (size_t)bh * S * DHEAD;
    const unsigned short* Kp = Kb + (size_t)bh * S * DHEAD;
    const unsigned short* Vp = Vt + (size_t)bh * S * DHEAD;   // (64, S) layout

    int lane = threadIdx.x & 31;
    int n = lane & 15, g = lane >> 4;

    // Q fragments (16 rows x 64 d) as two A-frags (K=32 each)
    FragBF aq[2];
#pragma unroll
    for (int ks = 0; ks < 2; ks++) {
        const unsigned short* qrow = Q + (size_t)(qb * 16 + n) * DHEAD + ks * 32;
        aq[ks].u4[0] = *reinterpret_cast<const uint4*>(qrow + g * 16);
        aq[ks].u4[1] = *reinterpret_cast<const uint4*>(qrow + g * 16 + 8);
    }

    v8f o[4];
    float mi[8], li[8];
#pragma unroll
    for (int t = 0; t < 4; t++)
#pragma unroll
        for (int r = 0; r < 8; r++) o[t][r] = 0.f;
#pragma unroll
    for (int r = 0; r < 8; r++) { mi[r] = -3.0e38f; li[r] = 0.f; }

    __shared__ float ldsP[16][32];
    int qend = qb * 16 + 15;

    for (int kbase = 0; kbase <= qend; kbase += 32) {
        // scores: two 16x16 tiles covering 32 keys
        v8f sc[2];
#pragma unroll
        for (int hn = 0; hn < 2; hn++) {
            v8f c;
#pragma unroll
            for (int r = 0; r < 8; r++) c[r] = 0.f;
#pragma unroll
            for (int ks = 0; ks < 2; ks++) {
                FragBF bk;
                const uint4* kp4 = reinterpret_cast<const uint4*>(
                    Kp + (size_t)(kbase + hn * 16 + n) * DHEAD + ks * 32 + g * 16);
                bk.u4[0] = kp4[0];
                bk.u4[1] = kp4[1];
                c = __builtin_amdgcn_wmma_f32_16x16x32_bf16(
                    false, aq[ks].v, false, bk.v, (short)0, c, false, false);
            }
            sc[hn] = c;
        }
        // scale + causal mask
#pragma unroll
        for (int hn = 0; hn < 2; hn++) {
#pragma unroll
            for (int r = 0; r < 8; r++) {
                int key = kbase + hn * 16 + n;
                int qm = qb * 16 + r + 8 * g;
                float v = sc[hn][r] * 0.125f;   // 1/sqrt(64)
                sc[hn][r] = (key <= qm) ? v : -3.0e38f;
            }
        }
        // online softmax stats (butterfly within 16-lane halves matches C layout)
        float alpha[8];
#pragma unroll
        for (int r = 0; r < 8; r++) {
            float mx = fmaxf(sc[0][r], sc[1][r]);
#pragma unroll
            for (int msk = 1; msk < 16; msk <<= 1)
                mx = fmaxf(mx, __shfl_xor(mx, msk, 32));
            float nm = fmaxf(mi[r], mx);
            float p0 = __expf(sc[0][r] - nm);
            float p1 = __expf(sc[1][r] - nm);
            sc[0][r] = p0; sc[1][r] = p1;
            float sum = p0 + p1;
#pragma unroll
            for (int msk = 1; msk < 16; msk <<= 1)
                sum += __shfl_xor(sum, msk, 32);
            alpha[r] = __expf(mi[r] - nm);
            li[r] = li[r] * alpha[r] + sum;
            mi[r] = nm;
        }
        // rescale accumulators
#pragma unroll
        for (int t = 0; t < 4; t++)
#pragma unroll
            for (int r = 0; r < 8; r++) o[t][r] *= alpha[r];

        // convert P (C layout) -> A fragment (16x32) through LDS
        __syncthreads();
#pragma unroll
        for (int r = 0; r < 8; r++) {
            ldsP[r + 8 * g][n]      = sc[0][r];
            ldsP[r + 8 * g][16 + n] = sc[1][r];
        }
        __syncthreads();
        FragBF ap;
#pragma unroll
        for (int w = 0; w < 8; w++) {
            int kk = g * 16 + w * 2;
            ap.s[2 * w + 0] = f2bf(ldsP[n][kk]);
            ap.s[2 * w + 1] = f2bf(ldsP[n][kk + 1]);
        }
        // O += P @ V using transposed V: lane d = t*16+n reads 16 contiguous keys
#pragma unroll
        for (int t = 0; t < 4; t++) {
            FragBF bv;
            const uint4* vp4 = reinterpret_cast<const uint4*>(
                Vp + (size_t)(t * 16 + n) * S + kbase + g * 16);
            bv.u4[0] = vp4[0];
            bv.u4[1] = vp4[1];
            o[t] = __builtin_amdgcn_wmma_f32_16x16x32_bf16(
                false, ap.v, false, bv.v, (short)0, o[t], false, false);
        }
    }

    // normalize + write bf16 (B*S, H) row-major
#pragma unroll
    for (int r = 0; r < 8; r++) {
        float inv = 1.0f / li[r];
        int srow = b * S + qb * 16 + r + 8 * g;
#pragma unroll
        for (int t = 0; t < 4; t++)
            outb[(size_t)srow * HID + h * DHEAD + t * 16 + n] = f2bf(o[t][r] * inv);
    }
}

// ---------------------------------------------------------------------------
extern "C" void kernel_launch(void* const* d_in, const int* in_sizes, int n_in,
                              void* d_out, int out_size, void* d_ws, size_t ws_size,
                              hipStream_t stream) {
    const float* x      = (const float*)d_in[0];
    const float* ln1_w  = (const float*)d_in[1];
    const float* ln2_w  = (const float*)d_in[2];
    const float* w_qkv  = (const float*)d_in[3];
    const float* w_out  = (const float*)d_in[4];
    const float* w_fc1  = (const float*)d_in[5];
    const float* w_fc2  = (const float*)d_in[6];
    float* out = (float*)d_out;

    const int B = 2, S = 2048, H = HID;
    const int M = B * S;           // 4096
    const int H3 = 3 * H, H4 = 4 * H;

    char* ws = (char*)d_ws;
    size_t off = 0;
    auto alloc = [&](size_t bytes) -> void* {
        void* p = ws + off;
        off = (off + bytes + 255) & ~(size_t)255;
        return p;
    };
    unsigned short* wqkv_t = (unsigned short*)alloc((size_t)H3 * H * 2);
    unsigned short* wout_t = (unsigned short*)alloc((size_t)H * H * 2);
    unsigned short* wfc1_t = (unsigned short*)alloc((size_t)H4 * H * 2);
    unsigned short* wfc2_t = (unsigned short*)alloc((size_t)H * H4 * 2);
    unsigned short* a1     = (unsigned short*)alloc((size_t)M * H * 2);
    float*          qkv    = (float*)alloc((size_t)M * H3 * 4);
    unsigned short* Qb     = (unsigned short*)alloc((size_t)M * H * 2);
    unsigned short* Kb     = (unsigned short*)alloc((size_t)M * H * 2);
    unsigned short* Vt     = (unsigned short*)alloc((size_t)M * H * 2);
    unsigned short* a2     = (unsigned short*)alloc((size_t)M * H * 2);
    float*          x1     = (float*)alloc((size_t)M * H * 4);
    unsigned short* a3     = (unsigned short*)alloc((size_t)M * H * 2);
    unsigned short* hbuf   = (unsigned short*)alloc((size_t)M * H4 * 2);

    // 1) weights -> bf16 transposed (NxK)
    wt_kernel<<<((size_t)H * H3 + 255) / 256, 256, 0, stream>>>(w_qkv, wqkv_t, H, H3);
    wt_kernel<<<((size_t)H * H  + 255) / 256, 256, 0, stream>>>(w_out, wout_t, H, H);
    wt_kernel<<<((size_t)H * H4 + 255) / 256, 256, 0, stream>>>(w_fc1, wfc1_t, H, H4);
    wt_kernel<<<((size_t)H4 * H + 255) / 256, 256, 0, stream>>>(w_fc2, wfc2_t, H4, H);

    // 2) ln1
    ln_kernel<<<M, 256, 0, stream>>>(x, ln1_w, a1);

    // 3) qkv = ln1 @ w_qkv
    gemm_bf16<0><<<dim3(H3 / BN, M / BM), 256, 0, stream>>>(a1, wqkv_t, qkv, nullptr, nullptr, M, H3, H);

    // 4) repack to per-head bf16 (V transposed)
    repack_qkv<<<((size_t)M * H + 255) / 256, 256, 0, stream>>>(qkv, Qb, Kb, Vt, S);

    // 5) causal attention
    attn_kernel<<<dim3(S / 16, NHEADS, B), 32, 0, stream>>>(Qb, Kb, Vt, a2, S);

    // 6) x1 = x + attn @ w_out
    gemm_bf16<1><<<dim3(H / BN, M / BM), 256, 0, stream>>>(a2, wout_t, x1, nullptr, x, M, H, H);

    // 7) ln2
    ln_kernel<<<M, 256, 0, stream>>>(x1, ln2_w, a3);

    // 8) h = gelu(ln2 @ w_fc1)  -> bf16
    gemm_bf16<2><<<dim3(H4 / BN, M / BM), 256, 0, stream>>>(a3, wfc1_t, nullptr, hbuf, nullptr, M, H4, H);

    // 9) out = x1 + h @ w_fc2
    gemm_bf16<1><<<dim3(H / BN, M / BM), 256, 0, stream>>>(hbuf, wfc2_t, out, nullptr, x1, M, H, H4);
}
